// LSTM_79078937854289
// MI455X (gfx1250) — matrix operand order repaired and checked by reading, not compiled, not run on hardware
//
#include <hip/hip_runtime.h>
#include <hip/hip_bf16.h>
#include <stdint.h>

typedef __attribute__((ext_vector_type(16))) __bf16 v16bf;
typedef __attribute__((ext_vector_type(8)))  float  v8f;

union Frag16 {
    v16bf v;
    uint4 u[2];
};

__device__ __forceinline__ uint16_t f32_to_bf16_rne(float f) {
    uint32_t u = __float_as_uint(f);
    uint32_t r = (u + 0x7FFFu + ((u >> 16) & 1u)) >> 16;
    return (uint16_t)r;
}

__device__ __forceinline__ float sigmoid_f(float x) {
    return 1.0f / (1.0f + __expf(-x));
}
__device__ __forceinline__ float tanh_f(float x) {
    float e = __expf(2.0f * x);
    return (e - 1.0f) / (e + 1.0f);
}

// ---------------------------------------------------------------------------
// Sizes
// ---------------------------------------------------------------------------
#define IN_DIM   512
#define HID      1024
#define G4       4096          // 4*HID
#define KCAT     1536          // IN_DIM + HID
#define BATCH    128
#define SEQ      512
#define OUT_DIM  128
// Fragment block: one 16x32 bf16 tile = 512 u16 = 1KB, stored lane-major
// (lane L's 32 bytes contiguous at block + 32*L), exactly the WMMA VGPR order.
#define FRAG_U16 512

// ---------------------------------------------------------------------------
// Prologue: pack W_cat = [W_ih | W_hh] into B-fragment-major bf16 layout.
// Layout: [nt = col tile 0..255][kb = 0..47][lane 0..31][slot 0..15]
// B-frag: lane = (col & 15) + 16*((kk>>3) & 1), slot = (kk & 7) + 8*((kk>>4) & 1)
// ---------------------------------------------------------------------------
__global__ void pack_wcat_frag(const float* __restrict__ Wih,
                               const float* __restrict__ Whh,
                               uint16_t* __restrict__ dst) {
    int i = blockIdx.x * blockDim.x + threadIdx.x;   // over 4096*1536
    if (i >= G4 * KCAT) return;
    const int block = i >> 9;           // which 1KB fragment
    const int within = i & 511;
    const int lane = within >> 4;
    const int slot = within & 15;
    const int nt = block / (KCAT / 32); // col tile
    const int kb = block % (KCAT / 32);
    const int n  = nt * 16 + (lane & 15);
    const int kk = ((slot >= 8) ? 16 : 0) + ((lane >= 16) ? 8 : 0) + (slot & 7);
    const int k  = kb * 32 + kk;
    float v = (k < IN_DIM) ? Wih[n * IN_DIM + k] : Whh[n * HID + (k - IN_DIM)];
    dst[i] = f32_to_bf16_rne(v);
}

// W_fc into B-fragment layout: [nt 0..7][kb 0..31][512]
__global__ void pack_wfc_frag(const float* __restrict__ Wfc,
                              uint16_t* __restrict__ dst) {
    int i = blockIdx.x * blockDim.x + threadIdx.x;   // over 128*1024
    if (i >= OUT_DIM * HID) return;
    const int block = i >> 9;
    const int within = i & 511;
    const int lane = within >> 4;
    const int slot = within & 15;
    const int nt = block / (HID / 32);
    const int kb = block % (HID / 32);
    const int n  = nt * 16 + (lane & 15);
    const int kk = ((slot >= 8) ? 16 : 0) + ((lane >= 16) ? 8 : 0) + (slot & 7);
    const int k  = kb * 32 + kk;
    dst[i] = f32_to_bf16_rne(Wfc[n * HID + k]);
}

// x into A-fragment layout: [t 0..511][m 0..7][kb 0..15][512]
// A-frag: lane = (row & 15) + 16*((kk>>3) & 1), slot = (kk & 7) + 8*((kk>>4) & 1)
// inverse: kk = 16*(slot>=8) + 8*(lane>=16) + (slot & 7)
__global__ void pack_x_frag(const float* __restrict__ x,
                            uint16_t* __restrict__ dst, int n_total) {
    int i = blockIdx.x * blockDim.x + threadIdx.x;
    int stride = gridDim.x * blockDim.x;
    for (; i < n_total; i += stride) {
        const int block = i >> 9;
        const int within = i & 511;
        const int lane = within >> 4;
        const int slot = within & 15;
        const int t  = block >> 7;          // /128
        const int rem = block & 127;
        const int m  = rem >> 4;
        const int kb = rem & 15;
        const int row = m * 16 + (lane & 15);
        const int kk  = ((slot >= 8) ? 16 : 0) + ((lane >= 16) ? 8 : 0) + (slot & 7);
        const int k   = kb * 32 + kk;
        dst[i] = f32_to_bf16_rne(x[(size_t)row * (SEQ * IN_DIM) + (size_t)t * IN_DIM + k]);
    }
}

// bias = b_ih + b_hh ; zero both h fragment buffers (bf16) and c (f32)
__global__ void init_state(const float* __restrict__ bih,
                           const float* __restrict__ bhh,
                           float* __restrict__ bias,
                           float* __restrict__ c,
                           uint16_t* __restrict__ h0,
                           uint16_t* __restrict__ h1) {
    int i = blockIdx.x * blockDim.x + threadIdx.x;   // 131072 threads
    if (i < G4) bias[i] = bih[i] + bhh[i];
    if (i < BATCH * HID) { c[i] = 0.0f; h0[i] = 0; h1[i] = 0; }
}

// ---------------------------------------------------------------------------
// Fully fused LSTM step (fragment-major operands), 2 batch-tiles per wave.
// Wave (p, jt) computes all four gate tiles for batch tiles m=2p,2p+1 and
// hidden cols jt*16..+15: each B (weight) fragment feeds TWO m-tiles, halving
// the dominant B read stream; i/f/g/o land in this wave's accumulators ->
// in-register cell math. h state kept in A-frag layout [m 0..7][kb 0..31][512].
// c state kept wave-linear: lane's 16 f32 values contiguous (coalesced).
// ---------------------------------------------------------------------------
__global__ __launch_bounds__(128) void lstm_step_fused(
        const uint16_t* __restrict__ xf,    // [512][8][16][512] A-frag bf16
        const uint16_t* __restrict__ h_in,  // [8][32][512] A-frag bf16 (read)
        const uint16_t* __restrict__ Wcat,  // [256][48][512] B-frag bf16
        const float*    __restrict__ bias,  // [4096]
        float*          __restrict__ c,     // [256 waves][32 lanes][16] f32
        uint16_t*       __restrict__ h_out, // [8][32][512] A-frag bf16 (write)
        int t) {
    const int wid  = (blockIdx.x * blockDim.x + threadIdx.x) >> 5; // 0..255
    const int lane = threadIdx.x & 31;
    const int p  = wid & 3;     // batch-tile pair: m = 2p, 2p+1
    const int jt = wid >> 2;    // hidden-slice tile (64)
    const int half = lane >> 4;
    const int lid  = lane & 15;
    const int j    = jt * 16 + lid;
    const int laneoff = lane * 16;                 // u16 offset of lane's 32B

    const uint16_t* xbase0 = xf + ((size_t)t * 128 + (size_t)(2 * p) * 16) * FRAG_U16 + laneoff;
    const uint16_t* xbase1 = xbase0 + 16 * FRAG_U16;          // next m-tile
    const uint16_t* hbase0 = h_in + ((size_t)(2 * p) * 32) * FRAG_U16 + laneoff;
    const uint16_t* hbase1 = hbase0 + 32 * FRAG_U16;
    const uint16_t* wbase[4];
    #pragma unroll
    for (int g = 0; g < 4; ++g)
        wbase[g] = Wcat + ((size_t)(g * 64 + jt) * (KCAT / 32)) * FRAG_U16 + laneoff;

    v8f acc[2][4] = {{{}, {}, {}, {}}, {{}, {}, {}, {}}};

    // ---- x_t region: 16 K-blocks ----
    #pragma unroll 2
    for (int kb = 0; kb < IN_DIM / 32; ++kb) {
        Frag16 a0, a1;
        a0.u[0] = *(const uint4*)(xbase0 + kb * FRAG_U16);
        a0.u[1] = *(const uint4*)(xbase0 + kb * FRAG_U16 + 8);
        a1.u[0] = *(const uint4*)(xbase1 + kb * FRAG_U16);
        a1.u[1] = *(const uint4*)(xbase1 + kb * FRAG_U16 + 8);
        #pragma unroll
        for (int g = 0; g < 4; ++g) {
            Frag16 b;
            b.u[0] = *(const uint4*)(wbase[g] + kb * FRAG_U16);
            b.u[1] = *(const uint4*)(wbase[g] + kb * FRAG_U16 + 8);
            acc[0][g] = __builtin_amdgcn_wmma_f32_16x16x32_bf16(
                          false, a0.v, false, b.v, (short)0, acc[0][g], false, false);
            acc[1][g] = __builtin_amdgcn_wmma_f32_16x16x32_bf16(
                          false, a1.v, false, b.v, (short)0, acc[1][g], false, false);
        }
    }

    // ---- h region: 32 K-blocks (W offset by 16 K-blocks) ----
    #pragma unroll 2
    for (int kb = 0; kb < HID / 32; ++kb) {
        Frag16 a0, a1;
        a0.u[0] = *(const uint4*)(hbase0 + kb * FRAG_U16);
        a0.u[1] = *(const uint4*)(hbase0 + kb * FRAG_U16 + 8);
        a1.u[0] = *(const uint4*)(hbase1 + kb * FRAG_U16);
        a1.u[1] = *(const uint4*)(hbase1 + kb * FRAG_U16 + 8);
        #pragma unroll
        for (int g = 0; g < 4; ++g) {
            Frag16 b;
            const uint16_t* bp = wbase[g] + (16 + kb) * FRAG_U16;
            b.u[0] = *(const uint4*)(bp);
            b.u[1] = *(const uint4*)(bp + 8);
            acc[0][g] = __builtin_amdgcn_wmma_f32_16x16x32_bf16(
                          false, a0.v, false, b.v, (short)0, acc[0][g], false, false);
            acc[1][g] = __builtin_amdgcn_wmma_f32_16x16x32_bf16(
                          false, a1.v, false, b.v, (short)0, acc[1][g], false, false);
        }
    }

    // ---- in-register LSTM cell epilogue; h written in A-frag layout ----
    const float bi = bias[0 * HID + j];
    const float bf = bias[1 * HID + j];
    const float bg = bias[2 * HID + j];
    const float bo = bias[3 * HID + j];

    // Fragment coords of hidden column j within its K-block:
    const int kbh   = j >> 5;                       // h K-block (0..31)
    const int kk    = j & 31;
    const int part  = kk >> 3;
    const int slotW = (kk & 7) + ((part & 2) ? 8 : 0);
    const int laneW = (part & 1) ? 16 : 0;          // + row (b & 15)

    float* cw = c + (((size_t)wid * 32) + lane) * 16;   // 16 f32, contiguous

    #pragma unroll
    for (int mi = 0; mi < 2; ++mi) {
        const int m = 2 * p + mi;
        uint16_t* hblk = h_out + ((size_t)m * 32 + kbh) * FRAG_U16 + slotW;
        #pragma unroll
        for (int r = 0; r < 8; ++r) {
            const int rowL = half * 8 + r;          // b & 15 (C/D: M = r + 8*half)
            const float ig = sigmoid_f(acc[mi][0][r] + bi);
            const float fg = sigmoid_f(acc[mi][1][r] + bf);
            const float gg = tanh_f  (acc[mi][2][r] + bg);
            const float og = sigmoid_f(acc[mi][3][r] + bo);
            const float cn = fg * cw[mi * 8 + r] + ig * gg;
            cw[mi * 8 + r] = cn;
            hblk[(laneW + rowL) * 16] = f32_to_bf16_rne(og * tanh_f(cn));
        }
    }
}

// ---------------------------------------------------------------------------
// Final FC: out[128,128] = h(A-frag bf16) x W_fc^T(B-frag bf16) + b_fc
// ---------------------------------------------------------------------------
__global__ __launch_bounds__(256) void fc_kernel(
        const uint16_t* __restrict__ h,     // [8][32][512] A-frag bf16
        const uint16_t* __restrict__ Wfc,   // [8][32][512] B-frag bf16
        const float*    __restrict__ bfc,   // [128]
        float*          __restrict__ out) { // [128, 128]
    const int wave = (blockIdx.x * blockDim.x + threadIdx.x) >> 5; // 0..63
    const int lane = threadIdx.x & 31;
    const int tm = wave & 7, tn = wave >> 3;
    const int half = lane >> 4, lid = lane & 15;
    const int laneoff = lane * 16;

    const uint16_t* abase = h   + ((size_t)tm * 32) * FRAG_U16 + laneoff;
    const uint16_t* bbase = Wfc + ((size_t)tn * 32) * FRAG_U16 + laneoff;

    v8f acc = {};
    #pragma unroll 4
    for (int kb = 0; kb < HID / 32; ++kb) {
        Frag16 a, b;
        a.u[0] = *(const uint4*)(abase + kb * FRAG_U16);
        a.u[1] = *(const uint4*)(abase + kb * FRAG_U16 + 8);
        b.u[0] = *(const uint4*)(bbase + kb * FRAG_U16);
        b.u[1] = *(const uint4*)(bbase + kb * FRAG_U16 + 8);
        acc = __builtin_amdgcn_wmma_f32_16x16x32_bf16(
                false, a.v, false, b.v, (short)0, acc, false, false);
    }

    const float bv = bfc[tn * 16 + lid];
    #pragma unroll
    for (int r = 0; r < 8; ++r)
        out[(size_t)(tm * 16 + half * 8 + r) * OUT_DIM + tn * 16 + lid] = acc[r] + bv;
}

// ---------------------------------------------------------------------------
// Launch
// ---------------------------------------------------------------------------
extern "C" void kernel_launch(void* const* d_in, const int* in_sizes, int n_in,
                              void* d_out, int out_size, void* d_ws, size_t ws_size,
                              hipStream_t stream) {
    const float* x    = (const float*)d_in[0];   // [128, 512, 512]
    const float* Wih  = (const float*)d_in[1];   // [4096, 512]
    const float* Whh  = (const float*)d_in[2];   // [4096, 1024]
    const float* bih  = (const float*)d_in[3];   // [4096]
    const float* bhh  = (const float*)d_in[4];   // [4096]
    const float* Wfc  = (const float*)d_in[5];   // [128, 1024]
    const float* bfc  = (const float*)d_in[6];   // [128]
    float* out = (float*)d_out;                  // [128, 128]

    // Workspace layout (all 1KB-aligned by construction)
    char* ws = (char*)d_ws;
    size_t off = 0;
    uint16_t* wcat_f = (uint16_t*)(ws + off); off += (size_t)G4 * KCAT * 2;            // 12.6 MB
    uint16_t* x_f    = (uint16_t*)(ws + off); off += (size_t)BATCH * SEQ * IN_DIM * 2; // 64 MB
    uint16_t* wfc_f  = (uint16_t*)(ws + off); off += (size_t)OUT_DIM * HID * 2;        // 256 KB
    float*    bias   = (float*)(ws + off);    off += (size_t)G4 * 4;                   // 16 KB
    float*    c_st   = (float*)(ws + off);    off += (size_t)BATCH * HID * 4;          // 512 KB
    uint16_t* h_buf0 = (uint16_t*)(ws + off); off += (size_t)BATCH * HID * 2;          // 256 KB
    uint16_t* h_buf1 = (uint16_t*)(ws + off); off += (size_t)BATCH * HID * 2;          // 256 KB

    // --- prologue: fragment-swizzled bf16 packing ---
    {
        const int n = G4 * KCAT;
        pack_wcat_frag<<<(n + 255) / 256, 256, 0, stream>>>(Wih, Whh, wcat_f);
    }
    {
        const int n = BATCH * SEQ * IN_DIM;   // 33.5M elems, grid-stride
        pack_x_frag<<<16384, 256, 0, stream>>>(x, x_f, n);
    }
    {
        const int n = OUT_DIM * HID;
        pack_wfc_frag<<<(n + 255) / 256, 256, 0, stream>>>(Wfc, wfc_f);
    }
    init_state<<<(BATCH * HID + 255) / 256, 256, 0, stream>>>(bih, bhh, bias, c_st,
                                                              h_buf0, h_buf1);

    // --- recurrence: 512 fused steps, h ping-pong ---
    uint16_t* hbufs[2] = { h_buf0, h_buf1 };
    for (int t = 0; t < SEQ; ++t) {
        const uint16_t* h_in  = hbufs[t & 1];
        uint16_t*       h_out = hbufs[(t + 1) & 1];
        // 256 waves = 64 blocks x 4 waves; one (2 m-tiles, jt, 4 gates) job/wave
        lstm_step_fused<<<64, 128, 0, stream>>>(x_f, h_in, wcat_f, bias,
                                                c_st, h_out, t);
    }

    // --- final FC (h after 512 steps lives in h_buf0) ---
    fc_kernel<<<8, 256, 0, stream>>>(hbufs[SEQ & 1], wfc_f, bfc, out);
}
